// SelfTransformer_83081847374175
// MI455X (gfx1250) — compile-verified, matrix operand order checked by reading, not run on hardware
//
#include <hip/hip_runtime.h>
#include <math.h>

typedef __bf16 bf16;
typedef __attribute__((ext_vector_type(16))) __bf16 v16bf;
typedef __attribute__((ext_vector_type(8)))  __bf16 v8bf;
typedef __attribute__((ext_vector_type(8)))  float   v8f;

#define DM   512
#define HN   8
#define DKH  64
#define DFF  2048
#define SEQ  512
#define NB   4
#define NL   6
#define VOC  32000
#define MTOK (NB*SEQ)   // 2048 rows of activations

// ---------------------------------------------------------------------------
// Load a 16x32 bf16 WMMA fragment (one lane's two 8-halfword runs)
// ---------------------------------------------------------------------------
__device__ __forceinline__ v16bf ld_frag(const bf16* p, int half)
{
    union { v16bf v; v8bf h[2]; } u;
    u.h[0] = *(const v8bf*)(p +      half * 8);
    u.h[1] = *(const v8bf*)(p + 16 + half * 8);
    return u.v;
}

struct Frag { v16bf a0, a1, b0, b1, b2, b3; };

__device__ __forceinline__ void load_frags(
    Frag& f, const bf16* ar0, const bf16* ar1,
    const bf16* br0, const bf16* br1, const bf16* br2, const bf16* br3,
    int k0, int half)
{
    f.a0 = ld_frag(ar0 + k0, half);
    f.a1 = ld_frag(ar1 + k0, half);
    f.b0 = ld_frag(br0 + k0, half);
    f.b1 = ld_frag(br1 + k0, half);
    f.b2 = ld_frag(br2 + k0, half);
    f.b3 = ld_frag(br3 + k0, half);
}

__device__ __forceinline__ void mma_frags(v8f acc[2][4], const Frag& f)
{
    acc[0][0] = __builtin_amdgcn_wmma_f32_16x16x32_bf16(false, f.a0, false, f.b0, (short)0, acc[0][0], false, false);
    acc[0][1] = __builtin_amdgcn_wmma_f32_16x16x32_bf16(false, f.a0, false, f.b1, (short)0, acc[0][1], false, false);
    acc[0][2] = __builtin_amdgcn_wmma_f32_16x16x32_bf16(false, f.a0, false, f.b2, (short)0, acc[0][2], false, false);
    acc[0][3] = __builtin_amdgcn_wmma_f32_16x16x32_bf16(false, f.a0, false, f.b3, (short)0, acc[0][3], false, false);
    acc[1][0] = __builtin_amdgcn_wmma_f32_16x16x32_bf16(false, f.a1, false, f.b0, (short)0, acc[1][0], false, false);
    acc[1][1] = __builtin_amdgcn_wmma_f32_16x16x32_bf16(false, f.a1, false, f.b1, (short)0, acc[1][1], false, false);
    acc[1][2] = __builtin_amdgcn_wmma_f32_16x16x32_bf16(false, f.a1, false, f.b2, (short)0, acc[1][2], false, false);
    acc[1][3] = __builtin_amdgcn_wmma_f32_16x16x32_bf16(false, f.a1, false, f.b3, (short)0, acc[1][3], false, false);
}

// ---------------------------------------------------------------------------
// Batched GEMM:  C[b] = alpha * A[b] @ Bt[b]^T (+bias) (ReLU opt)
// A:  bf16 row-major [M,K];  Bt: bf16 row-major [N,K]  (B^T)
// C:  f32 [M,ldc]; Cb optional bf16 shadow.
// Block (8 waves) computes a 256x64 C tile: wave w -> rows [w*32, w*32+32),
// all waves share the same 64-col B panel (WGP$/L2 reuse).
// Requires M%256==0, N%64==0, K%64==0.  Ping-pong pipelined K loop.
// ---------------------------------------------------------------------------
__global__ __launch_bounds__(256) void gemm_bt_wmma(
    const bf16* __restrict__ A,  long long sA,
    const bf16* __restrict__ Bt, long long sB,
    float* __restrict__ C,       long long sC, int ldc,
    bf16* __restrict__ Cb,       long long sCb,
    const float* __restrict__ bias,
    int M, int N, int K, float alpha, int relu)
{
    const int batch = blockIdx.z;
    A  += (long long)batch * sA;
    Bt += (long long)batch * sB;
    C  += (long long)batch * sC;
    if (Cb) Cb += (long long)batch * sCb;

    const int wave = threadIdx.x >> 5;
    const int lane = threadIdx.x & 31;
    const int half = lane >> 4;
    const int l16  = lane & 15;

    const int Nt  = N >> 6;
    const int mtB = blockIdx.x / Nt;     // 256-row block strip
    const int nt  = blockIdx.x % Nt;     // 64-col panel

    const int row0 = mtB * 256 + wave * 32;
    const bf16* ar0 = A + (long long)(row0      + l16) * K;
    const bf16* ar1 = A + (long long)(row0 + 16 + l16) * K;
    const bf16* br0 = Bt + (long long)(nt * 64      + l16) * K;
    const bf16* br1 = br0 + (long long)16 * K;
    const bf16* br2 = br0 + (long long)32 * K;
    const bf16* br3 = br0 + (long long)48 * K;

    v8f acc[2][4] = {};
    Frag f0, f1;

    load_frags(f0, ar0, ar1, br0, br1, br2, br3, 0, half);
    for (int k0 = 0; k0 < K; k0 += 64) {
        // issue next fragment loads before consuming current ones
        load_frags(f1, ar0, ar1, br0, br1, br2, br3, k0 + 32, half);
        if (k0 + 128 < K) {
            __builtin_prefetch(ar0 + k0 + 128, 0, 1);
            __builtin_prefetch(br0 + k0 + 128, 0, 1);
        }
        mma_frags(acc, f0);
        if (k0 + 64 < K)
            load_frags(f0, ar0, ar1, br0, br1, br2, br3, k0 + 64, half);
        mma_frags(acc, f1);
    }

#pragma unroll
    for (int sub = 0; sub < 2; ++sub) {
#pragma unroll
        for (int j = 0; j < 4; ++j) {
            const int col = nt * 64 + j * 16 + l16;
            const float bv = bias ? bias[col] : 0.0f;
#pragma unroll
            for (int r = 0; r < 8; ++r) {
                const int m = row0 + sub * 16 + half * 8 + r;
                float vv = acc[sub][j][r] * alpha + bv;
                if (relu) vv = fmaxf(vv, 0.0f);
                C[(long long)m * ldc + col] = vv;
                if (Cb) Cb[(long long)m * ldc + col] = (bf16)vv;
            }
        }
    }
}

// ---------------------------------------------------------------------------
// Block reductions (256 threads)
// ---------------------------------------------------------------------------
__device__ __forceinline__ float blk_max(float v, float* sm) {
    int t = threadIdx.x;
    sm[t] = v; __syncthreads();
    for (int s = 128; s > 0; s >>= 1) {
        if (t < s) sm[t] = fmaxf(sm[t], sm[t + s]);
        __syncthreads();
    }
    float r = sm[0]; __syncthreads();
    return r;
}
__device__ __forceinline__ float blk_sum(float v, float* sm) {
    int t = threadIdx.x;
    sm[t] = v; __syncthreads();
    for (int s = 128; s > 0; s >>= 1) {
        if (t < s) sm[t] += sm[t + s];
        __syncthreads();
    }
    float r = sm[0]; __syncthreads();
    return r;
}

// ---------------------------------------------------------------------------
// Embedding + positional: x = 2*(emb[tok]*512) + pe = emb*1024 + pe
// ---------------------------------------------------------------------------
__global__ __launch_bounds__(256) void embed_pe_k(
    const int* __restrict__ tok, const float* __restrict__ emb,
    const float* __restrict__ pe, float* __restrict__ x, bf16* __restrict__ xb,
    int n)
{
    int i = blockIdx.x * 256 + threadIdx.x;
    if (i >= n) return;
    int c  = i & (DM - 1);
    int bs = i >> 9;
    int s  = bs & (SEQ - 1);
    float v = emb[(long long)tok[bs] * DM + c] * 1024.0f + pe[(long long)s * DM + c];
    x[i] = v; xb[i] = (bf16)v;
}

// ---------------------------------------------------------------------------
// Weight fp32 [K,N] -> bf16 transposed [N,K]; batched over layers (blockIdx.z)
// ---------------------------------------------------------------------------
__global__ __launch_bounds__(256) void tpose_w(
    const float* __restrict__ W, bf16* __restrict__ Wt, int K, int N)
{
    long long base = (long long)blockIdx.z * K * N;
    long long i = (long long)blockIdx.x * 256 + threadIdx.x;
    if (i >= (long long)K * N) return;
    int k = (int)(i / N), n = (int)(i % N);
    Wt[base + (long long)n * K + k] = (bf16)W[base + i];
}

// ---------------------------------------------------------------------------
// Split projections into per-head layouts: qh/kh [B,H,S,dk], vh [B,H,dk,S]
// ---------------------------------------------------------------------------
__global__ __launch_bounds__(256) void split_heads_k(
    const float* __restrict__ q, const float* __restrict__ k,
    const float* __restrict__ v, bf16* __restrict__ qh,
    bf16* __restrict__ kh, bf16* __restrict__ vh)
{
    int i = blockIdx.x * 256 + threadIdx.x;          // 0 .. 3*MTOK*DM
    int which = i / (MTOK * DM);
    int rem   = i - which * (MTOK * DM);
    int c  = rem & (DM - 1);
    int bs = rem >> 9;
    int b = bs >> 9, s = bs & (SEQ - 1);
    int h = c >> 6,  d = c & 63;
    if (which == 0)
        qh[(((long long)b * HN + h) * SEQ + s) * DKH + d] = (bf16)q[rem];
    else if (which == 1)
        kh[(((long long)b * HN + h) * SEQ + s) * DKH + d] = (bf16)k[rem];
    else
        vh[(((long long)b * HN + h) * DKH + d) * SEQ + s] = (bf16)v[rem];
}

// ctxh f32 [B,H,S,dk] -> ctxb bf16 [B*S, DM]
__global__ __launch_bounds__(256) void merge_heads_k(
    const float* __restrict__ ctxh, bf16* __restrict__ ctxb)
{
    int i = blockIdx.x * 256 + threadIdx.x;          // MTOK*DM
    int c  = i & (DM - 1);
    int bs = i >> 9;
    int b = bs >> 9, s = bs & (SEQ - 1);
    int h = c >> 6,  d = c & 63;
    ctxb[i] = (bf16)ctxh[(((long long)b * HN + h) * SEQ + s) * DKH + d];
}

// ---------------------------------------------------------------------------
// Masked softmax over S=512 cols; one block per (b,h,q) row; bf16 out.
// mmode 0: src_mask [B,1,S]; mmode 1: tgt_mask [B,S,S]
// ---------------------------------------------------------------------------
__global__ __launch_bounds__(256) void softmax_mask_k(
    const float* __restrict__ sc, const int* __restrict__ mask, int mmode,
    bf16* __restrict__ attn)
{
    __shared__ float sm[256];
    const long long row = blockIdx.x;                // (b*H + h)*S + q
    const int q = (int)(row & (SEQ - 1));
    const int b = (int)(row / ((long long)HN * SEQ));
    const float* r = sc + row * SEQ;
    bf16* o = attn + row * SEQ;
    const int t = threadIdx.x;
    int i0 = t, i1 = t + 256;
    int m0 = (mmode == 0) ? mask[(long long)b * SEQ + i0]
                          : mask[((long long)b * SEQ + q) * SEQ + i0];
    int m1 = (mmode == 0) ? mask[(long long)b * SEQ + i1]
                          : mask[((long long)b * SEQ + q) * SEQ + i1];
    float x0 = m0 ? r[i0] : -1e9f;
    float x1 = m1 ? r[i1] : -1e9f;
    float mx = blk_max(fmaxf(x0, x1), sm);
    float e0 = __expf(x0 - mx), e1 = __expf(x1 - mx);
    float inv = 1.0f / blk_sum(e0 + e1, sm);
    o[i0] = (bf16)(e0 * inv);
    o[i1] = (bf16)(e1 * inv);
}

// ---------------------------------------------------------------------------
// out = LayerNorm(a + res) * g + be ; res optional; writes f32 + bf16 shadow
// ---------------------------------------------------------------------------
__global__ __launch_bounds__(256) void add_ln_k(
    const float* __restrict__ a, const float* __restrict__ res,
    const float* __restrict__ g, const float* __restrict__ be,
    float* __restrict__ out, bf16* __restrict__ outb)
{
    __shared__ float sm[256];
    const long long row = blockIdx.x;
    const int t = threadIdx.x;
    const float* ar = a + row * DM;
    float x0 = ar[t]       + (res ? res[row * DM + t]       : 0.0f);
    float x1 = ar[t + 256] + (res ? res[row * DM + t + 256] : 0.0f);
    float mu = blk_sum(x0 + x1, sm) * (1.0f / DM);
    float d0 = x0 - mu, d1 = x1 - mu;
    float var = blk_sum(d0 * d0 + d1 * d1, sm) * (1.0f / DM);
    float rs = rsqrtf(var + 1e-5f);
    float y0 = d0 * rs * g[t]       + be[t];
    float y1 = d1 * rs * g[t + 256] + be[t + 256];
    if (out)  { out[row * DM + t] = y0;        out[row * DM + t + 256] = y1; }
    if (outb) { outb[row * DM + t] = (bf16)y0; outb[row * DM + t + 256] = (bf16)y1; }
}

// ---------------------------------------------------------------------------
// In-place row log_softmax over V=32000 (one block per row)
// ---------------------------------------------------------------------------
__global__ __launch_bounds__(256) void log_softmax_k(float* __restrict__ x, int Vn)
{
    __shared__ float sm[256];
    float* r = x + (long long)blockIdx.x * Vn;
    const int t = threadIdx.x;
    float mx = -INFINITY;
    for (int i = t; i < Vn; i += 256) mx = fmaxf(mx, r[i]);
    mx = blk_max(mx, sm);
    float s = 0.0f;
    for (int i = t; i < Vn; i += 256) s += __expf(r[i] - mx);
    s = blk_sum(s, sm);
    float lse = mx + __logf(s);
    for (int i = t; i < Vn; i += 256) r[i] -= lse;
}

// ===========================================================================
extern "C" void kernel_launch(void* const* d_in, const int* in_sizes, int n_in,
                              void* d_out, int out_size, void* d_ws, size_t ws_size,
                              hipStream_t stream)
{
    (void)in_sizes; (void)n_in; (void)out_size; (void)ws_size;
    const int M = MTOK;

    const int*   src      = (const int*)  d_in[0];
    const int*   tgt      = (const int*)  d_in[1];
    const int*   src_mask = (const int*)  d_in[2];
    const int*   tgt_mask = (const int*)  d_in[3];
    const float* embed    = (const float*)d_in[4];
    const float* pe       = (const float*)d_in[5];
    const float* gen_w    = (const float*)d_in[6];
    const float* gen_b    = (const float*)d_in[7];

    struct P { const float *wq,*wk,*wv,*wo,*bo,*ln1g,*ln1b,*ln2g,*ln2b,
                           *fflng,*fflnb,*fc1w,*fc1b,*fc2w,*fc2b; };
    auto getP = [&](int base) {
        P p;
        p.wq   = (const float*)d_in[base + 0];
        p.wk   = (const float*)d_in[base + 1];
        p.wv   = (const float*)d_in[base + 2];
        p.wo   = (const float*)d_in[base + 3];
        p.bo   = (const float*)d_in[base + 4];
        p.ln1g = (const float*)d_in[base + 5];
        p.ln1b = (const float*)d_in[base + 6];
        p.ln2g = (const float*)d_in[base + 7];
        p.ln2b = (const float*)d_in[base + 8];
        p.fflng= (const float*)d_in[base + 9];
        p.fflnb= (const float*)d_in[base +10];
        p.fc1w = (const float*)d_in[base +11];
        p.fc1b = (const float*)d_in[base +12];
        p.fc2w = (const float*)d_in[base +13];
        p.fc2b = (const float*)d_in[base +14];
        return p;
    };
    P enc = getP(8), dslf = getP(23), dcrs = getP(38);

    // ---- workspace bump allocator -----------------------------------------
    size_t off = 0;
    auto alloc = [&](size_t bytes) -> void* {
        void* p = (char*)d_ws + off;
        off += (bytes + 255) & ~(size_t)255;
        return p;
    };
    struct WT { bf16 *wq, *wk, *wv, *wo, *fc1, *fc2; };
    auto allocWT = [&]() {
        WT w;
        w.wq  = (bf16*)alloc((size_t)NL * DM * DM  * 2);
        w.wk  = (bf16*)alloc((size_t)NL * DM * DM  * 2);
        w.wv  = (bf16*)alloc((size_t)NL * DM * DM  * 2);
        w.wo  = (bf16*)alloc((size_t)NL * DM * DM  * 2);
        w.fc1 = (bf16*)alloc((size_t)NL * DM * DFF * 2);
        w.fc2 = (bf16*)alloc((size_t)NL * DFF * DM * 2);
        return w;
    };
    WT wenc = allocWT(), wdslf = allocWT(), wdcrs = allocWT();
    bf16* genT = (bf16*)alloc((size_t)DM * VOC * 2);

    float* xf    = (float*)alloc((size_t)M * DM * 4);
    bf16*  xb    = (bf16*) alloc((size_t)M * DM * 2);
    float* qf    = (float*)alloc((size_t)M * DM * 4);
    float* kf    = (float*)alloc((size_t)M * DM * 4);
    float* vf    = (float*)alloc((size_t)M * DM * 4);
    bf16*  qh    = (bf16*) alloc((size_t)M * DM * 2);
    bf16*  kh    = (bf16*) alloc((size_t)M * DM * 2);
    bf16*  vh    = (bf16*) alloc((size_t)M * DM * 2);
    float* scores= (float*)alloc((size_t)NB * HN * SEQ * SEQ * 4);
    bf16*  attnb = (bf16*) alloc((size_t)NB * HN * SEQ * SEQ * 2);
    float* ctxh  = (float*)alloc((size_t)M * DM * 4);
    bf16*  ctxb  = (bf16*) alloc((size_t)M * DM * 2);
    float* aout  = (float*)alloc((size_t)M * DM * 4);
    float* x1f   = (float*)alloc((size_t)M * DM * 4);
    bf16*  x1b   = (bf16*) alloc((size_t)M * DM * 2);
    float* ffinf = (float*)alloc((size_t)M * DM * 4);
    bf16*  ffinb = (bf16*) alloc((size_t)M * DM * 2);
    float* hf    = (float*)alloc((size_t)M * DFF * 4);
    bf16*  hb    = (bf16*) alloc((size_t)M * DFF * 2);
    float* ffv   = (float*)alloc((size_t)M * DM * 4);
    float* memf  = (float*)alloc((size_t)M * DM * 4);
    bf16*  memb  = (bf16*) alloc((size_t)M * DM * 2);

    // ---- convert + transpose all weights to bf16 [N,K] --------------------
    auto conv = [&](const float* W, bf16* Wt, int K, int N, int batches) {
        long long total = (long long)K * N;
        dim3 g((unsigned)((total + 255) / 256), 1, batches);
        tpose_w<<<g, 256, 0, stream>>>(W, Wt, K, N);
    };
    auto convSet = [&](const P& p, const WT& w) {
        conv(p.wq,   w.wq,  DM, DM, NL);
        conv(p.wk,   w.wk,  DM, DM, NL);
        conv(p.wv,   w.wv,  DM, DM, NL);
        conv(p.wo,   w.wo,  DM, DM, NL);
        conv(p.fc1w, w.fc1, DM, DFF, NL);
        conv(p.fc2w, w.fc2, DFF, DM, NL);
    };
    convSet(enc, wenc); convSet(dslf, wdslf); convSet(dcrs, wdcrs);
    conv(gen_w, genT, DM, VOC, 1);

    // ---- GEMM launcher (block tile 256x64) --------------------------------
    auto gemm = [&](const bf16* A, long long sA, const bf16* Bt, long long sB,
                    float* C, long long sC, int ldc, bf16* Cb, long long sCb,
                    const float* bias, int m, int n, int k,
                    float alpha, int relu, int batches) {
        dim3 grid((unsigned)((m / 256) * (n / 64)), 1, batches);
        gemm_bt_wmma<<<grid, 256, 0, stream>>>(A, sA, Bt, sB, C, sC, ldc,
                                               Cb, sCb, bias, m, n, k, alpha, relu);
    };

    // ---- one attention+FFN sublayer --------------------------------------
    auto sublayer = [&](const P& p, const WT& w, int l,
                        const bf16* kvb, const int* mask, int mmode) {
        const long long DD = (long long)DM * DM;
        const bf16* wqT  = w.wq  + l * DD;
        const bf16* wkT  = w.wk  + l * DD;
        const bf16* wvT  = w.wv  + l * DD;
        const bf16* woT  = w.wo  + l * DD;
        const bf16* fc1T = w.fc1 + (long long)l * DM * DFF;
        const bf16* fc2T = w.fc2 + (long long)l * DFF * DM;

        gemm(xb,  0, wqT, 0, qf, 0, DM, nullptr, 0, nullptr, M, DM, DM, 1.f, 0, 1);
        gemm(kvb, 0, wkT, 0, kf, 0, DM, nullptr, 0, nullptr, M, DM, DM, 1.f, 0, 1);
        gemm(kvb, 0, wvT, 0, vf, 0, DM, nullptr, 0, nullptr, M, DM, DM, 1.f, 0, 1);
        split_heads_k<<<(3 * M * DM) / 256, 256, 0, stream>>>(qf, kf, vf, qh, kh, vh);

        // scores = Q @ K^T / sqrt(dk)   (32 batched GEMMs, M=N=512, K=64)
        gemm(qh, (long long)SEQ * DKH, kh, (long long)SEQ * DKH,
             scores, (long long)SEQ * SEQ, SEQ, nullptr, 0, nullptr,
             SEQ, SEQ, DKH, 0.125f, 0, NB * HN);
        softmax_mask_k<<<NB * HN * SEQ, 256, 0, stream>>>(scores, mask, mmode, attnb);

        // ctx = attn @ V   (M=512, N=64, K=512)
        gemm(attnb, (long long)SEQ * SEQ, vh, (long long)DKH * SEQ,
             ctxh, (long long)SEQ * DKH, DKH, nullptr, 0, nullptr,
             SEQ, DKH, SEQ, 1.f, 0, NB * HN);
        merge_heads_k<<<(M * DM) / 256, 256, 0, stream>>>(ctxh, ctxb);

        gemm(ctxb, 0, woT, 0, aout, 0, DM, nullptr, 0, p.bo + l * DM,
             M, DM, DM, 1.f, 0, 1);
        add_ln_k<<<M, 256, 0, stream>>>(aout, xf, p.ln1g + l * DM, p.ln1b + l * DM,
                                        x1f, x1b);
        add_ln_k<<<M, 256, 0, stream>>>(x1f, nullptr, p.fflng + l * DM,
                                        p.fflnb + l * DM, ffinf, ffinb);
        gemm(ffinb, 0, fc1T, 0, hf, 0, DFF, hb, 0, p.fc1b + l * DFF,
             M, DFF, DM, 1.f, 1, 1);
        gemm(hb, 0, fc2T, 0, ffv, 0, DM, nullptr, 0, p.fc2b + l * DM,
             M, DM, DFF, 1.f, 0, 1);
        add_ln_k<<<M, 256, 0, stream>>>(ffv, x1f, p.ln2g + l * DM, p.ln2b + l * DM,
                                        xf, xb);
    };

    // ---- encoder ----------------------------------------------------------
    embed_pe_k<<<(M * DM) / 256, 256, 0, stream>>>(src, embed, pe, xf, xb, M * DM);
    for (int l = 0; l < NL; ++l) sublayer(enc, wenc, l, xb, src_mask, 0);
    hipMemcpyAsync(memf, xf, (size_t)M * DM * 4, hipMemcpyDeviceToDevice, stream);
    hipMemcpyAsync(memb, xb, (size_t)M * DM * 2, hipMemcpyDeviceToDevice, stream);

    // ---- decoder ----------------------------------------------------------
    embed_pe_k<<<(M * DM) / 256, 256, 0, stream>>>(tgt, embed, pe, xf, xb, M * DM);
    for (int l = 0; l < NL; ++l) {
        sublayer(dslf, wdslf, l, xb,   tgt_mask, 1);   // masked self-attn
        sublayer(dcrs, wdcrs, l, memb, src_mask, 0);   // cross-attn on memory
    }

    // ---- generator + log_softmax -----------------------------------------
    gemm(xb, 0, genT, 0, (float*)d_out, 0, VOC, nullptr, 0, gen_b,
         M, VOC, DM, 1.f, 0, 1);
    log_softmax_k<<<M, 256, 0, stream>>>((float*)d_out, VOC);
}
